// Network_48507360641682
// MI455X (gfx1250) — matrix operand order repaired and checked by reading, not compiled, not run on hardware
//
#include <hip/hip_runtime.h>
#include <hip/hip_bf16.h>
#include <math.h>

// CDNA5 wave32 WMMA vector types
typedef __attribute__((ext_vector_type(16))) __bf16 v16bf;
typedef __attribute__((ext_vector_type(8)))  float  v8f;

#define B_TOTAL   8192
#define K_DIM     4096
#define N1        50      // level width
#define N1_PAD    64      // padded to 4 WMMA N-tiles
#define KT_COUNT  (K_DIM / 32)   // 128 K-tiles of 32
#define ROWS_WG   128
#define LDS_STRIDE 52

// ---------------------------------------------------------------------------
// Prep: repack W1 [4096,50] f32 -> bf16 in CDNA5 B-matrix (32x16 per tile)
// lane layout: lane L holds column (L&15), K = (L>>4)*8 + e (e<8), 16 + (L>>4)*8 + (e-8) (e>=8).
// Tile (kt,nt) occupies 512 bf16 at ((kt*4+nt)*512); lane block of 16 each.
// Columns >= 50 are zero-padded.
// ---------------------------------------------------------------------------
__global__ __launch_bounds__(256) void repack_w1_bf16(
    const float* __restrict__ W1, __bf16* __restrict__ Bp) {
  int id   = blockIdx.x * 256 + threadIdx.x;  // 0 .. 262143
  int e    = id & 15;
  int lane = (id >> 4) & 31;
  int nt   = (id >> 9) & 3;
  int kt   = id >> 11;
  int col  = nt * 16 + (lane & 15);
  int k    = kt * 32 + (lane >> 4) * 8 + (e < 8 ? e : 8 + e);
  float v  = (col < N1) ? W1[k * N1 + col] : 0.0f;
  Bp[id]   = (__bf16)v;
}

// ---------------------------------------------------------------------------
// Fused forward: level = state@W1+b1 (bf16 WMMA, f32 acc), then the small
// tail (concat, W2, Wh, Wo, sigmoid) per-row in VALU.
// Grid: 64 WGs x 256 thr (8 waves). Wave w handles rows wgBase + w*16 ...
// ---------------------------------------------------------------------------
__global__ __launch_bounds__(256) void rnn_fused_fwd(
    const float* __restrict__ state,   // [8192,4096]
    const float* __restrict__ hidden,  // [8192,50]
    const __bf16* __restrict__ Bp,     // packed W1, 128*4*512 bf16
    const float* __restrict__ b1,      // [50]
    const float* __restrict__ W2,      // [100,32]
    const float* __restrict__ b2,      // [32]
    const float* __restrict__ Wh,      // [32,50]
    const float* __restrict__ bh,      // [50]
    const float* __restrict__ Wo,      // [50,1]
    const float* __restrict__ bo,      // [1]
    float* __restrict__ out)           // [8192] output ++ [8192*50] hidden_new
{
  __shared__ float lvl[ROWS_WG * LDS_STRIDE];

  const int tid  = threadIdx.x;
  const int wave = tid >> 5;
  const int lane = tid & 31;
  const int hl   = lane >> 4;        // lane half (K-half selector)
  const int nlo  = lane & 15;        // row (for A) / column (for B,C)

  const long wgBase = (long)blockIdx.x * ROWS_WG;
  const long rowA   = wgBase + wave * 16 + nlo;   // A: lanes 0-15 & 16-31 share rows
  const float* __restrict__ aRow = state + rowA * K_DIM + hl * 8;
  const __bf16* __restrict__ bLane = Bp + (size_t)lane * 16;

  v8f acc[4] = {v8f{}, v8f{}, v8f{}, v8f{}};

#pragma unroll 2
  for (int kt = 0; kt < KT_COUNT; ++kt) {
    // ---- A tile: 16 f32 per lane -> bf16.  K = kt*32 + hl*8 + {0..7, 16..23}
    const float* ap = aRow + kt * 32;
    const float4 s0 = *(const float4*)(ap + 0);
    const float4 s1 = *(const float4*)(ap + 4);
    const float4 s2 = *(const float4*)(ap + 16);
    const float4 s3 = *(const float4*)(ap + 20);
    v16bf a;
    a[0]=(__bf16)s0.x; a[1]=(__bf16)s0.y; a[2]=(__bf16)s0.z; a[3]=(__bf16)s0.w;
    a[4]=(__bf16)s1.x; a[5]=(__bf16)s1.y; a[6]=(__bf16)s1.z; a[7]=(__bf16)s1.w;
    a[8]=(__bf16)s2.x; a[9]=(__bf16)s2.y; a[10]=(__bf16)s2.z; a[11]=(__bf16)s2.w;
    a[12]=(__bf16)s3.x; a[13]=(__bf16)s3.y; a[14]=(__bf16)s3.z; a[15]=(__bf16)s3.w;

    // ---- B tiles (pre-packed, 32B-aligned vector loads) + 4 WMMAs
    const size_t tb = (size_t)(kt * 4) * 512;
    const v16bf b0 = *(const v16bf*)(bLane + tb);
    const v16bf b1v = *(const v16bf*)(bLane + tb + 512);
    const v16bf b2v = *(const v16bf*)(bLane + tb + 1024);
    const v16bf b3v = *(const v16bf*)(bLane + tb + 1536);

    acc[0] = __builtin_amdgcn_wmma_f32_16x16x32_bf16(false, a, false, b0,
                                                     (short)0, acc[0], false, false);
    acc[1] = __builtin_amdgcn_wmma_f32_16x16x32_bf16(false, a, false, b1v,
                                                     (short)0, acc[1], false, false);
    acc[2] = __builtin_amdgcn_wmma_f32_16x16x32_bf16(false, a, false, b2v,
                                                     (short)0, acc[2], false, false);
    acc[3] = __builtin_amdgcn_wmma_f32_16x16x32_bf16(false, a, false, b3v,
                                                     (short)0, acc[3], false, false);
  }

  // ---- bias + spill level tile to LDS.
  // C/D layout: lane L -> N = L&15 ; VGPR r -> M = r + 8*(L>>4)
  const int lrBase = wave * 16 + hl * 8;
#pragma unroll
  for (int nt = 0; nt < 4; ++nt) {
    const int col = nt * 16 + nlo;
    if (col < N1) {
      const float bb = b1[col];
#pragma unroll
      for (int r = 0; r < 8; ++r) {
        lvl[(lrBase + r) * LDS_STRIDE + col] = acc[nt][r] + bb;
      }
    }
  }
  __syncthreads();

  // ---- per-row tail: one thread per row (threads 128..255 idle here)
  if (tid < ROWS_WG) {
    const long row = wgBase + tid;
    const float* __restrict__ hrow = hidden + row * N1;

    // level2 = concat(level, hidden) @ W2 + b2      [32]
    float l2[32];
#pragma unroll
    for (int j = 0; j < 32; ++j) l2[j] = b2[j];
    for (int k = 0; k < N1; ++k) {
      const float ck = lvl[tid * LDS_STRIDE + k];
      const float* w = W2 + k * 32;
#pragma unroll
      for (int j = 0; j < 32; ++j) l2[j] = fmaf(ck, w[j], l2[j]);
    }
    for (int k = 0; k < N1; ++k) {
      const float ck = hrow[k];
      const float* w = W2 + (N1 + k) * 32;
#pragma unroll
      for (int j = 0; j < 32; ++j) l2[j] = fmaf(ck, w[j], l2[j]);
    }

    // hidden_new = level2 @ Wh + bh  [50]; output = sigmoid(hidden_new @ Wo + bo)
    float o = bo[0];
    for (int j = 0; j < N1; ++j) {
      float s = bh[j];
#pragma unroll
      for (int k = 0; k < 32; ++k) s = fmaf(l2[k], Wh[k * N1 + j], s);
      out[B_TOTAL + row * N1 + j] = s;       // hidden_new, flat after output
      o = fmaf(s, Wo[j], o);
    }
    out[row] = 1.0f / (1.0f + __expf(-o));   // sigmoid
  }
}

// ---------------------------------------------------------------------------
extern "C" void kernel_launch(void* const* d_in, const int* in_sizes, int n_in,
                              void* d_out, int out_size, void* d_ws, size_t ws_size,
                              hipStream_t stream) {
  const float* state  = (const float*)d_in[0];
  const float* hidden = (const float*)d_in[1];
  const float* W1     = (const float*)d_in[2];
  const float* b1     = (const float*)d_in[3];
  const float* W2     = (const float*)d_in[4];
  const float* b2     = (const float*)d_in[5];
  const float* Wh     = (const float*)d_in[6];
  const float* bh     = (const float*)d_in[7];
  const float* Wo     = (const float*)d_in[8];
  const float* bo     = (const float*)d_in[9];
  float* out          = (float*)d_out;

  // 512 KB of scratch for the bf16-repacked W1 (B-matrix lane layout).
  __bf16* Bp = (__bf16*)d_ws;

  // Repack W1: 128 kt * 4 nt * 512 elements = 262144 threads.
  repack_w1_bf16<<<1024, 256, 0, stream>>>(W1, Bp);

  // Fused forward: 8192 rows / 128 rows-per-WG = 64 workgroups.
  rnn_fused_fwd<<<64, 256, 0, stream>>>(state, hidden, Bp, b1, W2, b2,
                                        Wh, bh, Wo, bo, out);
}
